// MinGRU_33586644255262
// MI455X (gfx1250) — matrix operand order repaired and checked by reading, not compiled, not run on hardware
//
#include <hip/hip_runtime.h>
#include <hip/hip_bf16.h>

// ---------------------------------------------------------------------------
// MinGRU for MI455X (gfx1250):
//   pass 0: f32 -> bf16 pre-convert of x / Wz / Wh (bandwidth-bound, one time)
//   pass 1: dual bf16 WMMA GEMM (z and h~), async global->LDS double-buffered
//   pass 2-4: chunked parallel scan of h_t = (1-z)h_{t-1} + z*h~
// ---------------------------------------------------------------------------

typedef __attribute__((ext_vector_type(16))) __bf16 v16bf;
typedef __attribute__((ext_vector_type(8)))  __bf16 v8bf;
typedef __attribute__((ext_vector_type(8)))  float  v8f;
typedef int v4i_ __attribute__((vector_size(16)));   // payload type for async b128

#define B_      8
#define T_      4096
#define D_      1024
#define CHUNKS  32
#define CLEN    128            // T_/CHUNKS

#define BM      64             // block tile M (rows of flattened B*T)
#define BN      128            // block tile N (output features)
#define BK      32             // K step (WMMA bf16 K)
#define NKSTEP  (D_ / BK)
#define LDSKB   40             // bf16 LDS row stride: 80B rows -> 16B-aligned frags

#if defined(__AMDGCN__) && __has_builtin(__builtin_amdgcn_global_load_async_to_lds_b128)
#define USE_ASYNC_LDS 1
#else
#define USE_ASYNC_LDS 0
#endif

__device__ __forceinline__ unsigned bfbits(float f) {
  unsigned u = __builtin_bit_cast(unsigned, f);
  u += 0x7FFFu + ((u >> 16) & 1u);          // round-to-nearest-even
  return u >> 16;
}

__device__ __forceinline__ void async_copy_b128(const __bf16* g, __bf16* l) {
#if USE_ASYNC_LDS
  typedef __attribute__((address_space(1))) v4i_* gp_t;   // global b128 ptr
  typedef __attribute__((address_space(3))) v4i_* lp_t;   // LDS b128 ptr
  __builtin_amdgcn_global_load_async_to_lds_b128((gp_t)g, (lp_t)l, 0, 0);
#else
  *(uint4*)l = *(const uint4*)g;
#endif
}

__device__ __forceinline__ void wait_async_copies() {
#if USE_ASYNC_LDS
#if __has_builtin(__builtin_amdgcn_s_wait_asynccnt)
  __builtin_amdgcn_s_wait_asynccnt(0);
#else
  asm volatile("s_wait_asynccnt 0x0" ::: "memory");
#endif
#endif
}

// Build a v16bf fragment from two 16B-aligned LDS slices (2x ds_load_b128).
__device__ __forceinline__ v16bf frag16(const __bf16* p0, const __bf16* p1) {
  v8bf lo = *(const v8bf*)p0;
  v8bf hi = *(const v8bf*)p1;
  return __builtin_shufflevector(lo, hi, 0, 1, 2, 3, 4, 5, 6, 7,
                                 8, 9, 10, 11, 12, 13, 14, 15);
}

// ---------------------------------------------------------------------------
// Kernel 0: vectorized f32 -> bf16 conversion (8 elements / thread).
// ---------------------------------------------------------------------------
__global__ __launch_bounds__(256)
void cvt_f32_bf16_kernel(const float* __restrict__ src,
                         __bf16* __restrict__ dst, int n8) {
  const int i = blockIdx.x * 256 + threadIdx.x;
  if (i >= n8) return;
  const float4 a = ((const float4*)src)[2 * i + 0];
  const float4 b = ((const float4*)src)[2 * i + 1];
  uint4 o;
  o.x = bfbits(a.x) | (bfbits(a.y) << 16);
  o.y = bfbits(a.z) | (bfbits(a.w) << 16);
  o.z = bfbits(b.x) | (bfbits(b.y) << 16);
  o.w = bfbits(b.z) | (bfbits(b.w) << 16);
  ((uint4*)dst)[i] = o;
}

// ---------------------------------------------------------------------------
// Kernel 1: fused dual GEMM, bf16 inputs, f32 accumulate.
//   z  = sigmoid(x @ Wz^T + bz) -> zbuf (workspace)
//   h~ =          x @ Wh^T + bh -> out  (overwritten in place by pass 4)
// 256 threads = 8 waves arranged 2(M) x 4(N); each wave owns a 32x32 tile of
// BOTH outputs (A fragments reused) => 8 v_wmma per K step.
// Double-buffered LDS, async global->LDS copies overlap WMMA.
// ---------------------------------------------------------------------------
__global__ __launch_bounds__(256)
void mingru_gemm_kernel(const __bf16* __restrict__ xbf,
                        const __bf16* __restrict__ wzbf,
                        const float*  __restrict__ bz,
                        const __bf16* __restrict__ whbf,
                        const float*  __restrict__ bh,
                        float* __restrict__ zbuf,
                        float* __restrict__ htbuf) {
  __shared__ __bf16 smem[2][(BM + 2 * BN) * LDSKB];   // 2 x 25 KB

  const int tid  = threadIdx.x;
  const int lane = tid & 31;
  const int wave = tid >> 5;
  const int wm   = wave & 1;        // wave M index (0..1)
  const int wn   = wave >> 1;       // wave N index (0..3)

  const int m0 = blockIdx.y * BM;
  const int n0 = blockIdx.x * BN;

  const int half = lane >> 4;       // 0: lanes 0-15, 1: lanes 16-31
  const int l15  = lane & 15;
  const int akb  = half * 8;        // A-frag K sub-base  (ISA 16x32 layout)
  const int bkb  = half * 8;        // B-frag K sub-base  (ISA 32x16 layout)

  // staging coordinates: tile row r, 8-element (16B) chunk c
  const int xr = tid >> 2;                  // 0..63
  const int xc = (tid & 3) * 8;             // 0,8,16,24

  v8f accZ[2][2], accH[2][2];
#pragma unroll
  for (int i = 0; i < 2; ++i)
#pragma unroll
    for (int j = 0; j < 2; ++j) { accZ[i][j] = (v8f){}; accH[i][j] = (v8f){}; }

  // ---- issue async copies for K-step k0 into buffer bb -------------------
  auto issue = [&](int bb, int k0) {
    __bf16* xs  = &smem[bb][0];
    __bf16* wzs = &smem[bb][BM * LDSKB];
    __bf16* whs = &smem[bb][(BM + BN) * LDSKB];
    async_copy_b128(xbf + (size_t)(m0 + xr) * D_ + k0 + xc,
                    xs + xr * LDSKB + xc);
#pragma unroll
    for (int j = 0; j < 2; ++j) {
      const int idx = tid + j * 256;
      const int r   = idx >> 2;             // 0..127
      const int c   = (idx & 3) * 8;
      async_copy_b128(wzbf + (size_t)(n0 + r) * D_ + k0 + c,
                      wzs + r * LDSKB + c);
      async_copy_b128(whbf + (size_t)(n0 + r) * D_ + k0 + c,
                      whs + r * LDSKB + c);
    }
  };

  issue(0, 0);

  for (int kk = 0; kk < NKSTEP; ++kk) {
    const int cur = kk & 1;
    wait_async_copies();        // my copies done (RAW)
    __syncthreads();            // everyone's copies done; prev readers done (WAR)
    if (kk + 1 < NKSTEP) issue(cur ^ 1, (kk + 1) * BK);

    const __bf16* xs  = &smem[cur][0];
    const __bf16* wzs = &smem[cur][BM * LDSKB];
    const __bf16* whs = &smem[cur][(BM + BN) * LDSKB];

    // A fragments: rows wm*32 + mi*16 + l15; K slices {akb..+7, akb+16..+23}
    v16bf A[2];
#pragma unroll
    for (int mi = 0; mi < 2; ++mi) {
      const __bf16* rp = xs + (wm * 32 + mi * 16 + l15) * LDSKB;
      A[mi] = frag16(rp + akb, rp + akb + 16);
    }
#pragma unroll
    for (int ni = 0; ni < 2; ++ni) {
      const int col = wn * 32 + ni * 16 + l15;   // output feature = B column
      const __bf16* zp = wzs + col * LDSKB + bkb;
      const __bf16* hp = whs + col * LDSKB + bkb;
      const v16bf Bz = frag16(zp, zp + 16);
      const v16bf Bh = frag16(hp, hp + 16);
#pragma unroll
      for (int mi = 0; mi < 2; ++mi) {
        accZ[mi][ni] = __builtin_amdgcn_wmma_f32_16x16x32_bf16(
            false, A[mi], false, Bz, (short)0, accZ[mi][ni], false, false);
        accH[mi][ni] = __builtin_amdgcn_wmma_f32_16x16x32_bf16(
            false, A[mi], false, Bh, (short)0, accH[mi][ni], false, false);
      }
    }
  }

  // ---- epilogue: bias, sigmoid(z), store -------------------------------
#pragma unroll
  for (int mi = 0; mi < 2; ++mi) {
#pragma unroll
    for (int ni = 0; ni < 2; ++ni) {
      const int n = n0 + wn * 32 + ni * 16 + l15;
      const float bzv = bz[n];
      const float bhv = bh[n];
#pragma unroll
      for (int r = 0; r < 8; ++r) {
        const int m = m0 + wm * 32 + mi * 16 + half * 8 + r;   // C/D layout
        const size_t o = (size_t)m * D_ + n;
        const float zv = accZ[mi][ni][r] + bzv;
        zbuf[o]  = 1.0f / (1.0f + __expf(-zv));
        htbuf[o] = accH[mi][ni][r] + bhv;
      }
    }
  }
}

// ---------------------------------------------------------------------------
// Kernel 2: per-chunk composition of h' = a*h + b, a = 1-z, b = z*h~.
// One thread per (b, d, chunk); d fastest => coalesced.
// ---------------------------------------------------------------------------
__global__ __launch_bounds__(256)
void mingru_chunk_reduce(const float* __restrict__ zbuf,
                         const float* __restrict__ htbuf,
                         float* __restrict__ cA,
                         float* __restrict__ cB) {
  const int g  = blockIdx.x * 256 + threadIdx.x;  // (b*CHUNKS + c)*D + d
  const int d  = g & (D_ - 1);
  const int bc = g >> 10;
  const int c  = bc & (CHUNKS - 1);
  const int b  = bc >> 5;

  size_t idx = ((size_t)b * T_ + (size_t)c * CLEN) * D_ + d;
  float A = 1.0f, Bv = 0.0f;
#pragma unroll 4
  for (int t = 0; t < CLEN; ++t) {
    if (t + 8 < CLEN) {
      __builtin_prefetch(zbuf  + idx + 8 * D_, 0, 1);   // global_prefetch_b8
      __builtin_prefetch(htbuf + idx + 8 * D_, 0, 1);
    }
    const float z  = zbuf[idx];
    const float ht = htbuf[idx];
    const float a  = 1.0f - z;
    Bv = fmaf(a, Bv, z * ht);
    A *= a;
    idx += D_;
  }
  cA[g] = A;
  cB[g] = Bv;
}

// ---------------------------------------------------------------------------
// Kernel 3: serial scan across the 32 chunks (tiny). One thread per (b, d).
// ---------------------------------------------------------------------------
__global__ __launch_bounds__(256)
void mingru_chunk_scan(const float* __restrict__ hprev,
                       const float* __restrict__ cA,
                       const float* __restrict__ cB,
                       float* __restrict__ hstart,
                       float* __restrict__ hlast) {
  const int g = blockIdx.x * 256 + threadIdx.x;   // b*D + d
  const int d = g & (D_ - 1);
  const int b = g >> 10;
  float h = hprev[g];
#pragma unroll
  for (int c = 0; c < CHUNKS; ++c) {
    const int o = (b * CHUNKS + c) * D_ + d;
    hstart[o] = h;
    h = fmaf(cA[o], h, cB[o]);
  }
  hlast[g] = h;
}

// ---------------------------------------------------------------------------
// Kernel 4: apply recurrence inside each chunk; overwrites h~ in d_out.
// ---------------------------------------------------------------------------
__global__ __launch_bounds__(256)
void mingru_apply(const float* __restrict__ zbuf,
                  const float* __restrict__ hstart,
                  float* __restrict__ out) {
  const int g  = blockIdx.x * 256 + threadIdx.x;  // (b*CHUNKS + c)*D + d
  const int d  = g & (D_ - 1);
  const int bc = g >> 10;
  const int c  = bc & (CHUNKS - 1);
  const int b  = bc >> 5;

  float h = hstart[g];
  size_t idx = ((size_t)b * T_ + (size_t)c * CLEN) * D_ + d;
#pragma unroll 4
  for (int t = 0; t < CLEN; ++t) {
    if (t + 8 < CLEN) {
      __builtin_prefetch(zbuf + idx + 8 * D_, 0, 1);
      __builtin_prefetch(out  + idx + 8 * D_, 0, 1);
    }
    const float z  = zbuf[idx];
    const float ht = out[idx];
    h = fmaf(z, ht - h, h);     // (1-z)*h + z*ht
    out[idx] = h;
    idx += D_;
  }
}

// ---------------------------------------------------------------------------
extern "C" void kernel_launch(void* const* d_in, const int* in_sizes, int n_in,
                              void* d_out, int out_size, void* d_ws, size_t ws_size,
                              hipStream_t stream) {
  const float* x     = (const float*)d_in[0];   // [B,T,D]
  const float* hprev = (const float*)d_in[1];   // [B,D]
  const float* Wz    = (const float*)d_in[2];   // [D,D]
  const float* bz    = (const float*)d_in[3];   // [D]
  const float* Wh    = (const float*)d_in[4];   // [D,D]
  const float* bh    = (const float*)d_in[5];   // [D]

  float* out   = (float*)d_out;                 // outputs [B,T,D]
  float* hlast = out + (size_t)B_ * T_ * D_;    // h_last  [B,D]

  const size_t BTD = (size_t)B_ * T_ * D_;
  const size_t DD  = (size_t)D_ * D_;

  // Workspace (floats/bf16): zbuf | xbf | wzbf | whbf | cA | cB | hstart
  float*  zbuf   = (float*)d_ws;                     // BTD f32
  __bf16* xbf    = (__bf16*)(zbuf + BTD);            // BTD bf16
  __bf16* wzbf   = xbf + BTD;                        // DD bf16
  __bf16* whbf   = wzbf + DD;                        // DD bf16
  float*  cA     = (float*)(whbf + DD);              // B*CHUNKS*D
  float*  cB     = cA + (size_t)B_ * CHUNKS * D_;
  float*  hstart = cB + (size_t)B_ * CHUNKS * D_;

  // Pass 0: pre-convert to bf16 (one-time, bandwidth-bound).
  cvt_f32_bf16_kernel<<<(int)(BTD / 8 / 256), 256, 0, stream>>>(x,  xbf,  (int)(BTD / 8));
  cvt_f32_bf16_kernel<<<(int)(DD  / 8 / 256), 256, 0, stream>>>(Wz, wzbf, (int)(DD / 8));
  cvt_f32_bf16_kernel<<<(int)(DD  / 8 / 256), 256, 0, stream>>>(Wh, whbf, (int)(DD / 8));

  // Pass 1: dual WMMA GEMM.
  dim3 g1(D_ / BN, (B_ * T_) / BM);   // 8 x 512 blocks
  mingru_gemm_kernel<<<g1, 256, 0, stream>>>(xbf, wzbf, bz, whbf, bh, zbuf, out);

  // Passes 2-4: chunked parallel scan.
  mingru_chunk_reduce<<<(B_ * CHUNKS * D_) / 256, 256, 0, stream>>>(zbuf, out, cA, cB);
  mingru_chunk_scan  <<<(B_ * D_) / 256,          256, 0, stream>>>(hprev, cA, cB, hstart, hlast);
  mingru_apply       <<<(B_ * CHUNKS * D_) / 256, 256, 0, stream>>>(zbuf, hstart, out);
}
